// MultiHeadAttention_49873160241396
// MI455X (gfx1250) — matrix operand order repaired and checked by reading, not compiled, not run on hardware
//
#include <hip/hip_runtime.h>
#include <cstdint>
#include <cstddef>

// Problem constants (B, L, D) = (8, 1024, 1024); H=16, DK=DV=64.
#define NB   8
#define NL   1024
#define ND   1024
#define NH   16
#define NDK  64
#define LNEPS 1e-5f

typedef __attribute__((ext_vector_type(16))) __bf16 v16bf;
typedef __attribute__((ext_vector_type(8)))  __bf16 v8bf;
typedef __attribute__((ext_vector_type(8)))  float  v8f;

// ---------------------------------------------------------------------------
// Fragment helpers for V_WMMA_F32_16X16X32_BF16 (wave32).
// A (16x32, MxK): lane l (l<16): row M=l, K = 0..15; lane l+16:
//   row l, K = 16..31. element i: K = base + i,
//   base = 16*(l>=16).
// B (32x16, KxN): lane l holds column N = l&15; element i: K = 16*(l>=16)+i.
// C/D (16x16 f32): lane l, vgpr r -> (M = 8*(l>=16)+r, N = l&15).
// ---------------------------------------------------------------------------
__device__ __forceinline__ v16bf cat8(v8bf lo, v8bf hi) {
  return __builtin_shufflevector(lo, hi, 0,1,2,3,4,5,6,7,8,9,10,11,12,13,14,15);
}

__device__ __forceinline__ v8bf cvt8(const float* __restrict__ p) {
  const float4* q = reinterpret_cast<const float4*>(p);
  float4 f0 = q[0];
  float4 f1 = q[1];
  v8bf r;
  r[0] = (__bf16)f0.x; r[1] = (__bf16)f0.y; r[2] = (__bf16)f0.z; r[3] = (__bf16)f0.w;
  r[4] = (__bf16)f1.x; r[5] = (__bf16)f1.y; r[6] = (__bf16)f1.z; r[7] = (__bf16)f1.w;
  return r;
}

// p points at element K = k_step + 16*(lane>=16) of this lane's row.
__device__ __forceinline__ v16bf a_frag_f32(const float* __restrict__ p) {
  return cat8(cvt8(p), cvt8(p + 8));
}
// p points at element K = k_step + 16*(lane>=16) of this lane's column.
__device__ __forceinline__ v16bf b_frag_f32(const float* __restrict__ p) {
  return cat8(cvt8(p), cvt8(p + 8));
}
__device__ __forceinline__ v16bf a_frag_bf16(const __bf16* __restrict__ p) {
  return cat8(*reinterpret_cast<const v8bf*>(p), *reinterpret_cast<const v8bf*>(p + 8));
}
__device__ __forceinline__ v16bf b_frag_bf16(const __bf16* __restrict__ p) {
  return cat8(*reinterpret_cast<const v8bf*>(p), *reinterpret_cast<const v8bf*>(p + 8));
}

__device__ __forceinline__ v8f wmma_bf16(v16bf a, v16bf b, v8f c) {
  // (neg_a, A, neg_b, B, c_mod, C, reuse_a, reuse_b)
  return __builtin_amdgcn_wmma_f32_16x16x32_bf16(false, a, false, b, (short)0, c,
                                                 false, false);
}

// Async global->LDS copy of 16 bytes (GLOBAL_LOAD_ASYNC_TO_LDS_B128, ASYNCcnt).
__device__ __forceinline__ void async_b128_to_lds(const float* gsrc, float* ldst) {
  const unsigned lds_off = (unsigned)(size_t)(void*)ldst;  // generic ptr low 32b = LDS offset
  asm volatile("global_load_async_to_lds_b128 %0, %1, off"
               :: "v"(lds_off), "v"(gsrc)
               : "memory");
}
__device__ __forceinline__ void wait_asynccnt0() {
  asm volatile("s_wait_asynccnt 0x0" ::: "memory");
}

// ---------------------------------------------------------------------------
// Generic GEMM: C[M,N] = (A[M,K] @ W[N,K]^T + bias[N]) * npm[M] * scale
//   M = 8192, N = 1024, K = 1024. Block = 256 thr (8 waves).
//   Block tile 256M x 64N; each wave does a 32M x 64N strip (2x4 WMMA tiles).
//   The shared 64x32 f32 W-tile per k-step is double-buffered in LDS via
//   async global->LDS copies; A fragments stream from global (L2-resident).
// MODE 0: store bf16 row-major [M][1024]        (Qh / Kh)
// MODE 1: store bf16 transposed V: [b][h][d][l] (VhT)
// MODE 2: store f32 row-major [M][1024]         (final projection)
// ---------------------------------------------------------------------------
#define WROW 36   // LDS row stride in floats (distinct banks for r=0..15, 16B aligned)

template<int MODE>
__global__ __launch_bounds__(256)
void gemm_xwT_kernel(const float* __restrict__ A, const float* __restrict__ W,
                     const float* __restrict__ bias, const float* __restrict__ npm,
                     float scale, void* __restrict__ outp) {
  __shared__ float wtile[2][64 * WROW];
  const int K = 1024;
  const int t     = threadIdx.x;
  const int wave  = t >> 5;
  const int lane  = t & 31;
  const int lhalf = lane >> 4;      // 0|1
  const int lmod  = lane & 15;
  const int m_base = blockIdx.x * 256 + wave * 32;
  const int n_base = blockIdx.y * 64;

  // Async-stage W[n_base..n_base+63][kk..kk+31] into wtile[buf]:
  // 512 b128 chunks, 2 per thread.
  const int srow0 = t >> 3;            // chunk 0: rows 0..31
  const int spart = (t & 7) * 4;       // float column within 32-wide tile
  const float* wsrc0 = W + (size_t)(n_base + srow0) * K + spart;
  const float* wsrc1 = W + (size_t)(n_base + srow0 + 32) * K + spart;

  const float* aptr0 = A + (size_t)(m_base + lmod) * K + lhalf * 16;
  const float* aptr1 = aptr0 + (size_t)16 * K;

  // Prologue: stage kk = 0 into buffer 0.
  async_b128_to_lds(wsrc0, &wtile[0][srow0 * WROW + spart]);
  async_b128_to_lds(wsrc1, &wtile[0][(srow0 + 32) * WROW + spart]);

  v8f acc[2][4] = {};
  int buf = 0;
  for (int kk = 0; kk < K; kk += 32) {
    wait_asynccnt0();
    __syncthreads();                       // wtile[buf] visible to all waves
    if (kk + 32 < K) {
      const int nb = buf ^ 1;
      async_b128_to_lds(wsrc0 + kk + 32, &wtile[nb][srow0 * WROW + spart]);
      async_b128_to_lds(wsrc1 + kk + 32, &wtile[nb][(srow0 + 32) * WROW + spart]);
    }
    const v16bf a0 = a_frag_f32(aptr0 + kk);
    const v16bf a1 = a_frag_f32(aptr1 + kk);
#pragma unroll
    for (int j = 0; j < 4; ++j) {
      const float* lp = &wtile[buf][(j * 16 + lmod) * WROW + lhalf * 16];
      const v16bf b = cat8(cvt8(lp), cvt8(lp + 8));
      acc[0][j] = wmma_bf16(a0, b, acc[0][j]);
      acc[1][j] = wmma_bf16(a1, b, acc[1][j]);
    }
    __syncthreads();                       // all reads of wtile[buf] done
    buf ^= 1;
  }

#pragma unroll
  for (int i = 0; i < 2; ++i) {
    const int gm0 = m_base + i * 16 + lhalf * 8;
#pragma unroll
    for (int j = 0; j < 4; ++j) {
      const int gn = n_base + j * 16 + lmod;
      const float bs = bias[gn];
#pragma unroll
      for (int r = 0; r < 8; ++r) {
        const int gm = gm0 + r;
        const float val = (acc[i][j][r] + bs) * npm[gm] * scale;
        if (MODE == 0) {
          reinterpret_cast<__bf16*>(outp)[(size_t)gm * 1024 + gn] = (__bf16)val;
        } else if (MODE == 1) {
          const int bb = gm >> 10, li = gm & 1023;
          const int h  = gn >> 6,  d  = gn & 63;
          reinterpret_cast<__bf16*>(outp)[((size_t)((bb * NH + h) * NDK + d) << 10) | li] =
              (__bf16)val;
        } else {
          reinterpret_cast<float*>(outp)[(size_t)gm * 1024 + gn] = val;
        }
      }
    }
  }
}

// ---------------------------------------------------------------------------
// Scores: per (b,h): S[1024,1024] = Qh_bh[1024,64] @ Kh_bh[1024,64]^T
// (1/sqrt(DK) folded into Qh). Raw scores go straight to the attn output area.
// ---------------------------------------------------------------------------
__global__ __launch_bounds__(256)
void scores_kernel(const unsigned short* __restrict__ Qh_,
                   const unsigned short* __restrict__ Kh_,
                   float* __restrict__ S) {
  const __bf16* Qh = reinterpret_cast<const __bf16*>(Qh_);
  const __bf16* Kh = reinterpret_cast<const __bf16*>(Kh_);
  const int wave  = threadIdx.x >> 5;
  const int lane  = threadIdx.x & 31;
  const int lhalf = lane >> 4;
  const int lmod  = lane & 15;
  const int bh = blockIdx.z;                   // 0..127
  const int b  = bh >> 4, h = bh & 15;
  const int q_base = blockIdx.x * 128 + wave * 16;
  const int n_base = blockIdx.y * 64;

  const __bf16* qrow = Qh + ((size_t)(b * NL + q_base + lmod) << 10) + h * NDK + lhalf * 16;
  v8f acc[4] = {};
#pragma unroll
  for (int kk = 0; kk < NDK; kk += 32) {
    v16bf a = a_frag_bf16(qrow + kk);
#pragma unroll
    for (int j = 0; j < 4; ++j) {
      const __bf16* kptr =
          Kh + ((size_t)(b * NL + n_base + j * 16 + lmod) << 10) + h * NDK + kk + lhalf * 16;
      acc[j] = wmma_bf16(a, b_frag_bf16(kptr), acc[j]);
    }
  }

  float* sbase = S + ((size_t)bh << 20);
  const int gq0 = q_base + lhalf * 8;
#pragma unroll
  for (int j = 0; j < 4; ++j) {
    const int gk = n_base + j * 16 + lmod;
#pragma unroll
    for (int r = 0; r < 8; ++r)
      sbase[((size_t)(gq0 + r) << 10) + gk] = acc[j][r];
  }
}

// ---------------------------------------------------------------------------
// Masked softmax in place over rows of attn.  One block per (b,h,q) row.
// ---------------------------------------------------------------------------
__global__ __launch_bounds__(256)
void softmax_kernel(float* __restrict__ attn, const int* __restrict__ mask) {
  __shared__ float sred[256];
  const int row = blockIdx.x;                  // 0 .. B*H*L-1
  const int b   = row >> 14;
  const int qi  = row & 1023;
  float* p = attn + ((size_t)row << 10);
  const int* mp = mask + (((size_t)(b * NL + qi)) << 10);
  const int t = threadIdx.x;

  float vals[4];
  float mx = -INFINITY;
#pragma unroll
  for (int i = 0; i < 4; ++i) {
    const int c = t + i * 256;
    float e = p[c];
    if (mp[c] != 0) e = -INFINITY;
    vals[i] = e;
    mx = fmaxf(mx, e);
  }
  sred[t] = mx; __syncthreads();
  for (int s = 128; s > 0; s >>= 1) {
    if (t < s) sred[t] = fmaxf(sred[t], sred[t + s]);
    __syncthreads();
  }
  mx = sred[0]; __syncthreads();

  float sum = 0.f;
#pragma unroll
  for (int i = 0; i < 4; ++i) { vals[i] = __expf(vals[i] - mx); sum += vals[i]; }
  sred[t] = sum; __syncthreads();
  for (int s = 128; s > 0; s >>= 1) {
    if (t < s) sred[t] += sred[t + s];
    __syncthreads();
  }
  const float inv = 1.0f / sred[0];
#pragma unroll
  for (int i = 0; i < 4; ++i) p[t + i * 256] = vals[i] * inv;
}

// ---------------------------------------------------------------------------
// attn @ V: per (b,h): X2[b*L + q][h*64 + d] = sum_k attn[bh][q][k] * V[k][d]
// A fragments converted f32->bf16 on load; B from transposed bf16 VhT.
// ---------------------------------------------------------------------------
__global__ __launch_bounds__(256)
void av_kernel(const float* __restrict__ attn, const unsigned short* __restrict__ VhT_,
               float* __restrict__ X2) {
  const __bf16* VhT = reinterpret_cast<const __bf16*>(VhT_);
  const int wave  = threadIdx.x >> 5;
  const int lane  = threadIdx.x & 31;
  const int lhalf = lane >> 4;
  const int lmod  = lane & 15;
  const int bh = blockIdx.z;
  const int b  = bh >> 4, h = bh & 15;
  const int q_base = blockIdx.x * 128 + wave * 16;

  const float* arow = attn + ((((size_t)bh << 10) + q_base + lmod) << 10) + lhalf * 16;
  v8f acc[4] = {};
  for (int kk = 0; kk < NL; kk += 32) {
    v16bf a = a_frag_f32(arow + kk);
#pragma unroll
    for (int j = 0; j < 4; ++j) {
      const __bf16* vptr = VhT + ((size_t)(bh * NDK + j * 16 + lmod) << 10) + kk + lhalf * 16;
      acc[j] = wmma_bf16(a, b_frag_bf16(vptr), acc[j]);
    }
  }

  const int gq0 = q_base + lhalf * 8;
#pragma unroll
  for (int j = 0; j < 4; ++j) {
    const int col = h * NDK + j * 16 + lmod;
#pragma unroll
    for (int r = 0; r < 8; ++r)
      X2[((size_t)(b * NL + gq0 + r) << 10) + col] = acc[j][r];
  }
}

// ---------------------------------------------------------------------------
// Residual add + LayerNorm.  One block per row of 1024.
// ---------------------------------------------------------------------------
__global__ __launch_bounds__(256)
void ln_kernel(const float* __restrict__ Y, const float* __restrict__ resid,
               const float* __restrict__ gam, const float* __restrict__ bet,
               float* __restrict__ out) {
  __shared__ float sred[256];
  const int row = blockIdx.x;
  const float* yp = Y + ((size_t)row << 10);
  const float* rp = resid + ((size_t)row << 10);
  const int t = threadIdx.x;

  float x[4]; float s = 0.f;
#pragma unroll
  for (int i = 0; i < 4; ++i) { x[i] = yp[t + i * 256] + rp[t + i * 256]; s += x[i]; }
  sred[t] = s; __syncthreads();
  for (int st = 128; st > 0; st >>= 1) {
    if (t < st) sred[t] += sred[t + st];
    __syncthreads();
  }
  const float mu = sred[0] * (1.0f / 1024.0f);
  __syncthreads();

  float s2 = 0.f;
#pragma unroll
  for (int i = 0; i < 4; ++i) { const float d = x[i] - mu; s2 += d * d; }
  sred[t] = s2; __syncthreads();
  for (int st = 128; st > 0; st >>= 1) {
    if (t < st) sred[t] += sred[t + st];
    __syncthreads();
  }
  const float inv = rsqrtf(sred[0] * (1.0f / 1024.0f) + LNEPS);
#pragma unroll
  for (int i = 0; i < 4; ++i) {
    const int c = t + i * 256;
    out[((size_t)row << 10) + c] = (x[i] - mu) * inv * gam[c] + bet[c];
  }
}

// ---------------------------------------------------------------------------
// Launch.  Workspace layout (bytes):
//   [0,16M)   Qh  bf16        (later aliased by Y f32 spanning [0,32M))
//   [16M,32M) Kh  bf16
//   [32M,48M) VhT bf16
//   [48M,80M) X2  f32
// ---------------------------------------------------------------------------
extern "C" void kernel_launch(void* const* d_in, const int* in_sizes, int n_in,
                              void* d_out, int out_size, void* d_ws, size_t ws_size,
                              hipStream_t stream) {
  const float* q    = (const float*)d_in[0];
  const float* k    = (const float*)d_in[1];
  const float* v    = (const float*)d_in[2];
  const int*   mask = (const int*)d_in[3];
  const float* npm  = (const float*)d_in[4];
  const float* Wq   = (const float*)d_in[5];
  const float* bq   = (const float*)d_in[6];
  const float* Wk   = (const float*)d_in[7];
  const float* bk   = (const float*)d_in[8];
  const float* Wv   = (const float*)d_in[9];
  const float* bv   = (const float*)d_in[10];
  const float* Wf   = (const float*)d_in[11];
  const float* bf_  = (const float*)d_in[12];
  const float* lng  = (const float*)d_in[13];
  const float* lnb  = (const float*)d_in[14];

  float* out  = (float*)d_out;
  float* attn = out + (size_t)NB * NL * ND;          // 8,388,608 floats in

  char* ws = (char*)d_ws;
  unsigned short* Qh  = (unsigned short*)(ws);
  unsigned short* Kh  = (unsigned short*)(ws + (size_t)16 * 1024 * 1024);
  unsigned short* VhT = (unsigned short*)(ws + (size_t)32 * 1024 * 1024);
  float*          X2  = (float*)(ws + (size_t)48 * 1024 * 1024);
  float*          Y   = (float*)(ws);                // aliases dead Qh+Kh

  const dim3 blk(256);
  const dim3 gGemm(32, 16);                          // 8192/256 x 1024/64

  // Projections (1/sqrt(DK)=0.125 folded into Q).
  gemm_xwT_kernel<0><<<gGemm, blk, 0, stream>>>(q, Wq, bq, npm, 0.125f, (void*)Qh);
  gemm_xwT_kernel<0><<<gGemm, blk, 0, stream>>>(k, Wk, bk, npm, 1.0f,   (void*)Kh);
  gemm_xwT_kernel<1><<<gGemm, blk, 0, stream>>>(v, Wv, bv, npm, 1.0f,   (void*)VhT);

  // Scores -> attn region of d_out; masked softmax in place.
  scores_kernel<<<dim3(8, 16, NB * NH), blk, 0, stream>>>(Qh, Kh, attn);
  softmax_kernel<<<dim3(NB * NH * NL), blk, 0, stream>>>(attn, mask);

  // attn @ V -> X2 (f32 [8192][1024], head-major columns).
  av_kernel<<<dim3(8, 1, NB * NH), blk, 0, stream>>>(attn, VhT, X2);

  // Final projection (+bias, *npm) -> Y, then residual + LayerNorm -> out.
  gemm_xwT_kernel<2><<<gGemm, blk, 0, stream>>>(X2, Wf, bf_, npm, 1.0f, (void*)Y);
  ln_kernel<<<dim3(NB * NL), blk, 0, stream>>>(Y, q, lng, lnb, out);
}
